// MANY2MANY_LSTM_82995948028298
// MI455X (gfx1250) — matrix operand order repaired and checked by reading, not compiled, not run on hardware
//
#include <hip/hip_runtime.h>

#define Hh   2048
#define Gg   8192   // 4*H
#define Tt   1024
#define Cc   10
#define NBLK 64     // persistent blocks for recurrent kernel (64 x 256 = 512 waves)

typedef __attribute__((ext_vector_type(16))) __bf16 v16bf;
typedef __attribute__((ext_vector_type(8)))  float  v8f;

union Frag {
  v16bf v;
  unsigned short u[16];
  uint4 q[2];
};

// Native hardware convert (RNE) instead of manual bit-twiddling.
__device__ __forceinline__ unsigned short f2bf(float f) {
  union { __bf16 b; unsigned short u; } cv;
  cv.b = (__bf16)f;
  return cv.u;
}

__device__ __forceinline__ float sigf(float x) { return 1.0f / (1.0f + expf(-x)); }

__device__ __forceinline__ void cvt_store4(unsigned short* __restrict__ d,
                                           const float* __restrict__ s, int qi) {
  float4 f = ((const float4*)s)[qi];
  unsigned lo = (unsigned)f2bf(f.x) | ((unsigned)f2bf(f.y) << 16);
  unsigned hi = (unsigned)f2bf(f.z) | ((unsigned)f2bf(f.w) << 16);
  ((uint2*)d)[qi] = make_uint2(lo, hi);
}

// ---------------------------------------------------------------------------
// Kernel 1: fp32 -> bf16 for W_hh1, W_ih1, x (all stay L2-resident afterwards),
// plus state/barrier init. 4 elements per thread.
//   quads: W_hh1 4,194,304 | W_ih1 4,194,304 | x 524,288  => 34,816 blocks.
// ---------------------------------------------------------------------------
#define NWQ 4194304   // (Gg*Hh)/4
#define NXQ 524288    // (Tt*Hh)/4

__global__ __launch_bounds__(256) void prep_kernel(
    const float* __restrict__ Whh, const float* __restrict__ Wih,
    const float* __restrict__ x,
    unsigned short* __restrict__ whh_bf, unsigned short* __restrict__ wih_bf,
    unsigned short* __restrict__ x_bf,
    float* __restrict__ h1, float* __restrict__ c1,
    unsigned short* __restrict__ h1_bf, unsigned* __restrict__ bar) {
  int idx = blockIdx.x * 256 + threadIdx.x;
  if (idx < NWQ) {
    cvt_store4(whh_bf, Whh, idx);
  } else if (idx < 2 * NWQ) {
    cvt_store4(wih_bf, Wih, idx - NWQ);
  } else if (idx < 2 * NWQ + NXQ) {
    cvt_store4(x_bf, x, idx - 2 * NWQ);
  }
  if (idx < Hh) { h1[idx] = 0.0f; c1[idx] = 0.0f; h1_bf[idx] = 0; }
  if (idx < 128) bar[idx] = 0u;                        // cnt @0, gen @64
}

// ---------------------------------------------------------------------------
// Kernel 2: Xg = X @ W_ih1^T + (b_ih1 + b_hh1), pure-bf16 WMMA GEMM.
// One wave: 16(M=t) x 64(N=gate) block. Inner loop: 10x b128 loads + 4 wmma.
// Grid: 1024 blocks x 256 threads = 8192 waves = (1024/16)*(8192/64).
// High occupancy hides the per-wmma load waits.
// ---------------------------------------------------------------------------
__global__ __launch_bounds__(256) void gemm_xw_kernel(
    const unsigned short* __restrict__ xb, const unsigned short* __restrict__ wb,
    const float* __restrict__ bih1, const float* __restrict__ bhh1,
    float* __restrict__ Xg) {
  const int lane = threadIdx.x & 31;
  const int w    = blockIdx.x * 8 + (threadIdx.x >> 5);   // 0..8191
  const int m_base = (w >> 7) * 16;                       // t tile
  const int n_base = (w & 127) * 64;                      // gate tile
  const int kc_a = (lane & 16) ? 16 : 0;                  // A: K 0-15 vs 16-31
  const int kc_b = (lane & 16) ? 16 : 0;                  // B: K 0-15 vs 16-31

  const unsigned short* pa = xb + (long)(m_base + (lane & 15)) * Hh + kc_a;
  const unsigned short* pb = wb + (long)(n_base + (lane & 15)) * Hh + kc_b;

  v8f acc[4] = {v8f{}, v8f{}, v8f{}, v8f{}};

  for (int k0 = 0; k0 < Hh; k0 += 32) {
    Frag a, b0, b1, b2, b3;
    a.q[0]  = *(const uint4*)(pa + k0);
    a.q[1]  = *(const uint4*)(pa + k0 + 8);
    b0.q[0] = *(const uint4*)(pb + k0);
    b0.q[1] = *(const uint4*)(pb + k0 + 8);
    b1.q[0] = *(const uint4*)(pb + 16 * Hh + k0);
    b1.q[1] = *(const uint4*)(pb + 16 * Hh + k0 + 8);
    b2.q[0] = *(const uint4*)(pb + 32 * Hh + k0);
    b2.q[1] = *(const uint4*)(pb + 32 * Hh + k0 + 8);
    b3.q[0] = *(const uint4*)(pb + 48 * Hh + k0);
    b3.q[1] = *(const uint4*)(pb + 48 * Hh + k0 + 8);
    acc[0] = __builtin_amdgcn_wmma_f32_16x16x32_bf16(false, a.v, false, b0.v, (short)0, acc[0], false, false);
    acc[1] = __builtin_amdgcn_wmma_f32_16x16x32_bf16(false, a.v, false, b1.v, (short)0, acc[1], false, false);
    acc[2] = __builtin_amdgcn_wmma_f32_16x16x32_bf16(false, a.v, false, b2.v, (short)0, acc[2], false, false);
    acc[3] = __builtin_amdgcn_wmma_f32_16x16x32_bf16(false, a.v, false, b3.v, (short)0, acc[3], false, false);
  }

  const int nlo  = lane & 15;
  const int moff = (lane & 16) ? 8 : 0;
#pragma unroll
  for (int q = 0; q < 4; ++q) {
    int n = n_base + q * 16 + nlo;
    float bb = bih1[n] + bhh1[n];
#pragma unroll
    for (int r = 0; r < 8; ++r) {
      int m = m_base + moff + r;                          // D: VGPR r <-> M=r / M=8+r
      Xg[(long)m * Gg + n] = acc[q][r] + bb;
    }
  }
}

// ---------------------------------------------------------------------------
// Grid barrier (sense-reversing, atomic counter + generation).
// ---------------------------------------------------------------------------
__device__ __forceinline__ void grid_barrier(unsigned* cnt, unsigned* gen) {
  __threadfence();
  __syncthreads();
  if (threadIdx.x == 0) {
    volatile unsigned* vgen = gen;
    unsigned g = *vgen;
    __threadfence();
    unsigned prev = atomicAdd(cnt, 1u);
    if (prev == (unsigned)(NBLK - 1)) {
      *cnt = 0;
      __threadfence();
      atomicAdd(gen, 1u);
    } else {
      while (*vgen == g) { __builtin_amdgcn_s_sleep(1); }
    }
  }
  __threadfence();
  __syncthreads();
}

// ---------------------------------------------------------------------------
// Kernel 3: persistent recurrent kernel. Per step:
//   (a) g = h1 @ W_hh1^T via WMMA: A = 16 weight rows (M), B = broadcast h1;
//       each of 512 waves owns one 16-row gate tile, 64 wmma over K=2048.
//       Double-buffered fragments: next k-step's loads issue before the
//       current wmma's wait -> L2 latency overlapped at low occupancy.
//   (b) barrier; fp32 gate math -> h1, c1, h1_bf16 (2048 lanes).
//   (c) barrier; block 0: LSTMCell2 (hidden=1) + Linear -> out[t*10..].
// W_hh1(bf16)=32MB + Xg=32MB stay L2-resident (192MB L2).
// ---------------------------------------------------------------------------
__device__ __forceinline__ void load_rfrag(Frag& a, Frag& b,
    const unsigned short* __restrict__ wrow,
    const unsigned short* __restrict__ hb, int k0) {
  a.q[0] = *(const uint4*)(wrow + k0);
  a.q[1] = *(const uint4*)(wrow + k0 + 8);
  b.q[0] = *(const uint4*)(hb + k0);
  b.q[1] = *(const uint4*)(hb + k0 + 8);
}

__global__ __launch_bounds__(256) void lstm_recurrent_kernel(
    const unsigned short* __restrict__ whh_bf, const float* __restrict__ Xg,
    const float* __restrict__ Wih2, const float* __restrict__ Whh2,
    const float* __restrict__ bih2, const float* __restrict__ bhh2,
    const float* __restrict__ Wlin, const float* __restrict__ blin,
    float* __restrict__ g, float* __restrict__ h1, float* __restrict__ c1,
    unsigned short* __restrict__ h1_bf, unsigned* __restrict__ bar,
    float* __restrict__ out) {
  const int lane = threadIdx.x & 31;
  const int wgid = blockIdx.x * 8 + (threadIdx.x >> 5);   // 0..511
  const int jrow = wgid * 16 + (lane & 15);               // this lane's gate row
  const int kc_a = (lane & 16) ? 16 : 0;
  const int kc_b = (lane & 16) ? 16 : 0;
  const int gid  = blockIdx.x * 256 + threadIdx.x;
  const unsigned short* wrow = whh_bf + (long)jrow * Hh + kc_a;
  const unsigned short* hb   = h1_bf + kc_b;
  unsigned* cnt = bar;
  unsigned* gen = bar + 64;

  float h2 = 0.0f, c2 = 0.0f;                             // cell-2 state (block0/tid0)
  __shared__ float sred[256];

  for (int t = 0; t < Tt; ++t) {
    // ---- (a) recurrent matvec, software-pipelined (double-buffered frags)
    v8f acc = v8f{};
    Frag a0, b0, a1, b1;
    load_rfrag(a0, b0, wrow, hb, 0);
    for (int k0 = 0; k0 < Hh - 64; k0 += 64) {
      load_rfrag(a1, b1, wrow, hb, k0 + 32);
      acc = __builtin_amdgcn_wmma_f32_16x16x32_bf16(false, a0.v, false, b0.v, (short)0, acc, false, false);
      load_rfrag(a0, b0, wrow, hb, k0 + 64);
      acc = __builtin_amdgcn_wmma_f32_16x16x32_bf16(false, a1.v, false, b1.v, (short)0, acc, false, false);
    }
    load_rfrag(a1, b1, wrow, hb, Hh - 32);
    acc = __builtin_amdgcn_wmma_f32_16x16x32_bf16(false, a0.v, false, b0.v, (short)0, acc, false, false);
    acc = __builtin_amdgcn_wmma_f32_16x16x32_bf16(false, a1.v, false, b1.v, (short)0, acc, false, false);

    if (lane == 0) {                                      // D column N=0: lanes 0 & 16
#pragma unroll
      for (int r = 0; r < 8; ++r) g[wgid * 16 + r] = acc[r];
    } else if (lane == 16) {
#pragma unroll
      for (int r = 0; r < 8; ++r) g[wgid * 16 + 8 + r] = acc[r];
    }
    grid_barrier(cnt, gen);

    // ---- (b) gate nonlinearities, state update (2048 elements)
    if (gid < Hh) {
      const float* xg = Xg + (long)t * Gg;                // already has b_ih1+b_hh1
      float iv = sigf (xg[0 * Hh + gid] + g[0 * Hh + gid]);
      float fv = sigf (xg[1 * Hh + gid] + g[1 * Hh + gid]);
      float gv = tanhf(xg[2 * Hh + gid] + g[2 * Hh + gid]);
      float ov = sigf (xg[3 * Hh + gid] + g[3 * Hh + gid]);
      float cn = fv * c1[gid] + iv * gv;
      c1[gid] = cn;
      float hn = ov * tanhf(cn);
      h1[gid] = hn;
      h1_bf[gid] = f2bf(hn);
    }
    grid_barrier(cnt, gen);

    // ---- (c) LSTMCell2 (hidden size 1) + Linear, block 0 only
    if (blockIdx.x == 0) {
      float p0 = 0.f, p1 = 0.f, p2 = 0.f, p3 = 0.f;
      for (int k = (int)threadIdx.x; k < Hh; k += 256) {
        float hv = h1[k];
        p0 += Wih2[0 * Hh + k] * hv;
        p1 += Wih2[1 * Hh + k] * hv;
        p2 += Wih2[2 * Hh + k] * hv;
        p3 += Wih2[3 * Hh + k] * hv;
      }
      float parts[4] = {p0, p1, p2, p3};
      float g2[4];
#pragma unroll
      for (int r = 0; r < 4; ++r) {
        __syncthreads();
        sred[threadIdx.x] = parts[r];
        __syncthreads();
        for (int s = 128; s > 0; s >>= 1) {
          if ((int)threadIdx.x < s) sred[threadIdx.x] += sred[threadIdx.x + s];
          __syncthreads();
        }
        g2[r] = sred[0];
      }
      if (threadIdx.x == 0) {
        float iv = g2[0] + bih2[0] + Whh2[0] * h2 + bhh2[0];
        float fv = g2[1] + bih2[1] + Whh2[1] * h2 + bhh2[1];
        float gv = g2[2] + bih2[2] + Whh2[2] * h2 + bhh2[2];
        float ov = g2[3] + bih2[3] + Whh2[3] * h2 + bhh2[3];
        c2 = sigf(fv) * c2 + sigf(iv) * tanhf(gv);
        h2 = sigf(ov) * tanhf(c2);
#pragma unroll
        for (int c = 0; c < Cc; ++c) out[t * Cc + c] = Wlin[c] * h2 + blin[c];
      }
    }
  }
}

// ---------------------------------------------------------------------------
extern "C" void kernel_launch(void* const* d_in, const int* in_sizes, int n_in,
                              void* d_out, int out_size, void* d_ws, size_t ws_size,
                              hipStream_t stream) {
  (void)in_sizes; (void)n_in; (void)out_size; (void)ws_size;
  const float* x    = (const float*)d_in[0];
  const float* Wih1 = (const float*)d_in[1];
  const float* Whh1 = (const float*)d_in[2];
  const float* bih1 = (const float*)d_in[3];
  const float* bhh1 = (const float*)d_in[4];
  const float* Wih2 = (const float*)d_in[5];
  const float* Whh2 = (const float*)d_in[6];
  const float* bih2 = (const float*)d_in[7];
  const float* bhh2 = (const float*)d_in[8];
  const float* Wlin = (const float*)d_in[9];
  const float* blin = (const float*)d_in[10];
  float* out = (float*)d_out;

  // Workspace layout (~100.1 MB total; all hot data L2-resident on 192MB L2)
  char* ws = (char*)d_ws;
  unsigned short* whh_bf = (unsigned short*)ws;                 // 32 MB
  unsigned short* wih_bf = (unsigned short*)(ws + (size_t)33554432);   // 32 MB
  unsigned short* x_bf   = (unsigned short*)(ws + (size_t)67108864);   // 4 MB
  float* Xg = (float*)(ws + (size_t)71303168);                  // 32 MB
  float* g  = (float*)(ws + (size_t)104857600);                 // 32 KB
  float* h1 = g + Gg;                                           // 8 KB
  float* c1 = h1 + Hh;                                          // 8 KB
  unsigned short* h1_bf = (unsigned short*)(c1 + Hh);           // 4 KB
  unsigned* bar = (unsigned*)(h1_bf + Hh);                      // 512 B

  prep_kernel<<<34816, 256, 0, stream>>>(Whh1, Wih1, x, whh_bf, wih_bf, x_bf,
                                         h1, c1, h1_bf, bar);
  gemm_xw_kernel<<<1024, 256, 0, stream>>>(x_bf, wih_bf, bih1, bhh1, Xg);
  lstm_recurrent_kernel<<<NBLK, 256, 0, stream>>>(whh_bf, Xg, Wih2, Whh2, bih2,
                                                  bhh2, Wlin, blin, g, h1, c1,
                                                  h1_bf, bar, out);
}
